// SequenceClassifier_32950989095501
// MI455X (gfx1250) — matrix operand order repaired and checked by reading, not compiled, not run on hardware
//
#include <hip/hip_runtime.h>
#include <cstdint>
#include <cstddef>

// ---------------------------------------------------------------------------
// 2-layer LSTM + classification + autoregressive generation for MI455X
// (gfx1250, wave32, WMMA).  Matmuls on v_wmma_f32_16x16x32_f16 with f32
// accumulation; cell state in f32.  Input GEMMs folded into the per-step
// K loop.  A-operand tiles staged into LDS with global_load_async_to_lds
// (double buffered, prefetched one k-step ahead); weight operand register
// double-buffered; all four A fragments hoisted ahead of the WMMA group.
// ---------------------------------------------------------------------------

typedef __attribute__((ext_vector_type(16))) _Float16 v16h;
typedef __attribute__((ext_vector_type(8)))  _Float16 v8h;
typedef __attribute__((ext_vector_type(8)))  float    v8f;

#define B_    64
#define S_    512
#define D_    128
#define H_    512
#define C_    32
#define PRED_ 640
#define G4H   (4*H_)
#define AROWP 40   // padded halfs per staged A row (80B -> bank-conflict free)

__device__ __forceinline__ float sigf(float x){ return 1.0f/(1.0f+__expf(-x)); }

__device__ __forceinline__ void wait_async0(){
  asm volatile("s_wait_asynccnt 0x0" ::: "memory");
}
// one lane copies 16B global -> LDS asynchronously (ASYNCcnt tracked)
__device__ __forceinline__ void async_g2l_b128(uint32_t ldsOff, const _Float16* g){
  asm volatile("global_load_async_to_lds_b128 %0, %1, off"
               :: "v"(ldsOff), "v"((uint64_t)(uintptr_t)g) : "memory");
}

// ------------------------- small utility kernels ---------------------------

__global__ void k_cvt(const float* __restrict__ src, _Float16* __restrict__ dst, long n){
  long i  = (long)blockIdx.x*blockDim.x + threadIdx.x;
  long st = (long)gridDim.x*blockDim.x;
  for(; i<n; i+=st) dst[i] = (_Float16)src[i];
}

__global__ void k_vadd(const float* __restrict__ a, const float* __restrict__ b,
                       float* __restrict__ d, int n){
  int i = blockIdx.x*blockDim.x + threadIdx.x;
  if(i<n) d[i] = a[i]+b[i];
}

// f16 copy of generated[:, S-1, :]  (seed of the autoregressive loop)
__global__ void k_yinit(const float* __restrict__ gen, _Float16* __restrict__ dst){
  int i = blockIdx.x*blockDim.x + threadIdx.x;  // B_*D_ threads
  if(i < B_*D_){
    int b = i / D_, c = i % D_;
    dst[i] = (_Float16)gen[(long)b*((long)PRED_*D_) + (long)(S_-1)*D_ + c];
  }
}

// ------------------------- WMMA fragment loaders ---------------------------
// A (16x32 f16, MxK):  lane<16: M=lane, elems0-15=K k0..k0+15
//                      lane>=16: M=lane-16, K k0+16..k0+31
__device__ __forceinline__ v16h load_a(const _Float16* __restrict__ base, long ld,
                                       int m0, int k0, int lane){
  int hf = lane >> 4, lr = lane & 15;
  const _Float16* p = base + (long)(m0+lr)*ld + k0 + 16*hf;
  union { v16h v; v8h h[2]; } u;
  u.h[0] = *(const v8h*)(p);
  u.h[1] = *(const v8h*)(p + 8);
  return u.v;
}

// B (32x16 f16, KxN) from row-major W(N,K): lane<16: N=lane, elems=K k0..k0+15;
// lane>=16: N=lane-16, K k0+16..k0+31  -> one contiguous 32B load per lane.
__device__ __forceinline__ v16h load_b(const _Float16* __restrict__ W, int K,
                                       int nrow, int k0, int lane){
  int hf = lane >> 4, lr = lane & 15;
  return *(const v16h*)(W + (long)(nrow+lr)*K + k0 + 16*hf);
}

// ------------------------------ LSTM step ----------------------------------
// gates = X@W1^T + Hprev@W2^T + bias ; i,f,g,o ; c'=f*c+i*g ; h'=o*tanh(c')
// grid = 32 blocks (16 h-columns each), block = 128 threads (4 waves, 1/gate).
// The 64x32 A-tile (shared by all 4 gate waves) is staged into LDS with
// async global->LDS copies, double buffered and prefetched one k-step ahead.
__global__ __launch_bounds__(128)
void k_lstm_step(const _Float16* __restrict__ X,  long ldx, int K1,
                 const _Float16* __restrict__ Hp, long ldh, int K2,
                 const _Float16* __restrict__ W1,
                 const _Float16* __restrict__ W2,
                 const float* __restrict__ bias,
                 float* __restrict__ Cst,
                 _Float16* __restrict__ Hout, long ldo)
{
  __shared__ __align__(32) _Float16 abuf[2][64][AROWP];
  __shared__ float glds[4][64][16];

  const int tid  = threadIdx.x;
  const int wv   = tid >> 5;        // gate id: 0=i 1=f 2=g 3=o
  const int lane = tid & 31;
  const int hf   = lane >> 4, lr = lane & 15;
  const int n0   = blockIdx.x * 16;        // h-column block
  const int grow = wv*H_ + n0;             // gate-row base in W

  const int n1 = K1 >> 5, n2 = K2 >> 5, nk = n1 + n2;

  // issue async stage of A tile for global k-step `ki` into buffer `bsel`
  auto stage = [&](int bsel, int ki){
    const _Float16* src; long ld; int k0;
    if(ki < n1){ src = X;  ld = ldx; k0 = ki*32; }
    else       { src = Hp; ld = ldh; k0 = (ki-n1)*32; }
    // 64 rows x 64B = 256 16B chunks; 128 threads -> 2 chunks each
    #pragma unroll
    for(int c = tid; c < 256; c += 128){
      int row = c >> 2, ch = c & 3;
      async_g2l_b128((uint32_t)(uintptr_t)(&abuf[bsel][row][ch*8]),
                     src + (long)row*ld + k0 + ch*8);
    }
  };
  // A fragment from the staged LDS tile
  auto afrag = [&](int bsel, int m)->v16h{
    const _Float16* r = &abuf[bsel][16*m + lr][16*hf];
    union { v16h v; v8h h[2]; } u;
    u.h[0] = *(const v8h*)(r);
    u.h[1] = *(const v8h*)(r + 8);
    return u.v;
  };
  auto bfrag = [&](int ki)->v16h{
    const _Float16* W; int K; int k0;
    if(ki < n1){ W = W1; K = K1; k0 = ki*32; }
    else       { W = W2; K = K2; k0 = (ki-n1)*32; }
    return *(const v16h*)(W + (long)(grow+lr)*K + k0 + 16*hf);
  };

  v8f acc[4] = {v8f{},v8f{},v8f{},v8f{}};

  stage(0, 0);
  v16h bcur = bfrag(0);
  for(int ki=0; ki<nk; ki++){
    wait_async0();            // our stage of buf[ki&1] landed in LDS
    __syncthreads();          // everyone's stage landed; prev readers done
    const bool more = (ki+1 < nk);
    if(more) stage((ki+1)&1, ki+1);     // prefetch next tile (async)
    v16h bnext = bcur;
    if(more) bnext = bfrag(ki+1);       // prefetch next weight fragment
    const int bsel = ki & 1;
    // hoist all four A fragments so the ds_load latency pipelines
    v16h a[4];
    #pragma unroll
    for(int m=0;m<4;m++) a[m] = afrag(bsel, m);
    #pragma unroll
    for(int m=0;m<4;m++)
      acc[m] = __builtin_amdgcn_wmma_f32_16x16x32_f16(false, a[m], false, bcur,
                                                      (short)0, acc[m], false, false);
    bcur = bnext;
  }

  const float bv = bias[grow + lr];
  #pragma unroll
  for(int m=0;m<4;m++)
    #pragma unroll
    for(int v=0;v<8;v++)
      glds[wv][16*m + 8*hf + v][lr] = acc[m][v] + bv;   // D layout: M=v+8*hf, N=lr

  __syncthreads();

  for(int e=tid; e<64*16; e+=128){
    int r = e >> 4, cc = e & 15;
    float gi = sigf (glds[0][r][cc]);
    float gf = sigf (glds[1][r][cc]);
    float gg = tanhf(glds[2][r][cc]);
    float go = sigf (glds[3][r][cc]);
    long  ci = (long)r*H_ + n0 + cc;
    float c  = gf*Cst[ci] + gi*gg;
    Cst[ci]  = c;
    Hout[(long)r*ldo + n0 + cc] = (_Float16)(go*tanhf(c));
  }
}

// ------------------------- generic WMMA GEMM -------------------------------
// out[r,c] = sum_k A[r,k]*W[c,k] + bias[c]
// f32 output index: oBase + (r/oDiv)*oB + (r%oDiv)*oS + c   (handles (B,640,D) remap)
// optional f16 secondary output outH[r*N+c].  Register double-buffered K loop.
__global__ __launch_bounds__(128)
void k_gemm(const _Float16* __restrict__ A, long ldA,
            const _Float16* __restrict__ W, int K,
            const float* __restrict__ bias,
            float* __restrict__ outF, long oDiv, long oB, long oS, long oBase,
            _Float16* __restrict__ outH,
            int M, int N, int nbN)
{
  const int bm = blockIdx.x / nbN, bn = blockIdx.x % nbN;
  const int wv = threadIdx.x >> 5, lane = threadIdx.x & 31;
  const int hf = lane >> 4, lr = lane & 15;
  const int n0 = bn*64 + wv*16;
  const int m0 = bm*64;
  if(n0 >= N) return;                        // wave-uniform exit (no barriers here)

  v8f acc[4] = {v8f{},v8f{},v8f{},v8f{}};
  v16h b0 = load_b(W, K, n0, 0, lane);
  v16h a0[4];
  #pragma unroll
  for(int m=0;m<4;m++) a0[m] = load_a(A, ldA, m0+16*m, 0, lane);

  for(int k0=0; k0<K; k0+=32){
    const bool more = (k0+32 < K);
    v16h b1 = b0; v16h a1[4];
    if(more){
      b1 = load_b(W, K, n0, k0+32, lane);
      #pragma unroll
      for(int m=0;m<4;m++) a1[m] = load_a(A, ldA, m0+16*m, k0+32, lane);
    }
    #pragma unroll
    for(int m=0;m<4;m++)
      acc[m] = __builtin_amdgcn_wmma_f32_16x16x32_f16(false, a0[m], false, b0,
                                                      (short)0, acc[m], false, false);
    b0 = b1;
    #pragma unroll
    for(int m=0;m<4;m++) a0[m] = a1[m];
  }

  const int   c  = n0 + lr;
  const float bv = bias ? bias[c] : 0.0f;
  #pragma unroll
  for(int m=0;m<4;m++){
    #pragma unroll
    for(int v=0;v<8;v++){
      int r = m0 + 16*m + 8*hf + v;
      if(r < M){
        float val = acc[m][v] + bv;
        long  oi  = oBase + ((long)r/oDiv)*oB + ((long)r%oDiv)*oS + c;
        outF[oi] = val;
        if(outH) outH[(long)r*N + c] = (_Float16)val;
      }
    }
  }
}

// ------------------------------- launcher ----------------------------------
extern "C" void kernel_launch(void* const* d_in, const int* in_sizes, int n_in,
                              void* d_out, int out_size, void* d_ws, size_t ws_size,
                              hipStream_t stream)
{
  (void)in_sizes; (void)n_in; (void)out_size; (void)ws_size;

  const float* x    = (const float*)d_in[0];
  const float* Wih0 = (const float*)d_in[1];
  const float* Whh0 = (const float*)d_in[2];
  const float* bih0 = (const float*)d_in[3];
  const float* bhh0 = (const float*)d_in[4];
  const float* Wih1 = (const float*)d_in[5];
  const float* Whh1 = (const float*)d_in[6];
  const float* bih1 = (const float*)d_in[7];
  const float* bhh1 = (const float*)d_in[8];
  const float* Wc   = (const float*)d_in[9];
  const float* bc   = (const float*)d_in[10];
  const float* Wg   = (const float*)d_in[11];
  const float* bg   = (const float*)d_in[12];
  // d_in[13] = predict_length (device scalar); fixed at 640 for this problem.

  float* outClass = (float*)d_out;            // 64 x 32
  float* outGen   = (float*)d_out + B_*C_;    // 64 x 640 x 128

  // --- workspace carve (~83 MB) ---
  char* p = (char*)d_ws;
  auto alloc = [&](size_t bytes)->char*{ char* r = p; p += (bytes + 255) & ~(size_t)255; return r; };
  _Float16* xh    = (_Float16*)alloc((size_t)B_*S_*D_*2);
  _Float16* w_ih0 = (_Float16*)alloc((size_t)G4H*D_*2);
  _Float16* w_hh0 = (_Float16*)alloc((size_t)G4H*H_*2);
  _Float16* w_ih1 = (_Float16*)alloc((size_t)G4H*H_*2);
  _Float16* w_hh1 = (_Float16*)alloc((size_t)G4H*H_*2);
  _Float16* wgh   = (_Float16*)alloc((size_t)D_*H_*2);
  _Float16* wch   = (_Float16*)alloc((size_t)C_*H_*2);
  float*    bias0 = (float*)alloc((size_t)G4H*4);
  float*    bias1 = (float*)alloc((size_t)G4H*4);
  _Float16* out0  = (_Float16*)alloc((size_t)B_*S_*H_*2);
  _Float16* out1  = (_Float16*)alloc((size_t)B_*S_*H_*2);
  float*    c0    = (float*)alloc((size_t)B_*H_*4);
  float*    c1    = (float*)alloc((size_t)B_*H_*4);
  _Float16* h0b[2] = {(_Float16*)alloc((size_t)B_*H_*2), (_Float16*)alloc((size_t)B_*H_*2)};
  _Float16* h1b[2] = {(_Float16*)alloc((size_t)B_*H_*2), (_Float16*)alloc((size_t)B_*H_*2)};
  _Float16* yb[2]  = {(_Float16*)alloc((size_t)B_*D_*2), (_Float16*)alloc((size_t)B_*D_*2)};

  auto cvt = [&](const float* s, _Float16* d, long n){
    int blocks = (int)((n + 255)/256); if(blocks > 8192) blocks = 8192;
    k_cvt<<<blocks,256,0,stream>>>(s,d,n);
  };
  cvt(x,    xh,    (long)B_*S_*D_);
  cvt(Wih0, w_ih0, (long)G4H*D_);
  cvt(Whh0, w_hh0, (long)G4H*H_);
  cvt(Wih1, w_ih1, (long)G4H*H_);
  cvt(Whh1, w_hh1, (long)G4H*H_);
  cvt(Wg,   wgh,   (long)D_*H_);
  cvt(Wc,   wch,   (long)C_*H_);
  k_vadd<<<8,256,0,stream>>>(bih0,bhh0,bias0,G4H);
  k_vadd<<<8,256,0,stream>>>(bih1,bhh1,bias1,G4H);
  hipMemsetAsync(c0, 0, (size_t)B_*H_*4, stream);
  hipMemsetAsync(c1, 0, (size_t)B_*H_*4, stream);

  // --- layer 0 scan (input GEMM folded into per-step K loop) ---
  for(int t=0; t<S_; t++){
    k_lstm_step<<<32,128,0,stream>>>(
        xh + (long)t*D_, (long)S_*D_, D_,
        t ? out0 + (long)(t-1)*H_ : (const _Float16*)nullptr, (long)S_*H_, t ? H_ : 0,
        w_ih0, w_hh0, bias0, c0,
        out0 + (long)t*H_, (long)S_*H_);
  }
  // --- layer 1 scan ---
  for(int t=0; t<S_; t++){
    k_lstm_step<<<32,128,0,stream>>>(
        out0 + (long)t*H_, (long)S_*H_, H_,
        t ? out1 + (long)(t-1)*H_ : (const _Float16*)nullptr, (long)S_*H_, t ? H_ : 0,
        w_ih1, w_hh1, bias1, c1,
        out1 + (long)t*H_, (long)S_*H_);
  }

  // --- gen_prompt: out1 (32768 x 512) @ Wg^T -> generated[:, :512, :] ---
  {
    int M = B_*S_, N = D_, nbN = (N+63)/64;
    int blocks = ((M+63)/64)*nbN;
    k_gemm<<<blocks,128,0,stream>>>(out1, (long)H_, wgh, H_, bg,
        outGen, 512L, (long)PRED_*D_, (long)D_, 0L, (_Float16*)nullptr, M, N, nbN);
  }
  // --- classification: out1[:, -1, :] @ Wc^T + bc ---
  k_gemm<<<1,128,0,stream>>>(out1 + (long)(S_-1)*H_, (long)S_*H_, wch, H_, bc,
      outClass, 1L, (long)C_, 0L, 0L, (_Float16*)nullptr, B_, C_, 1);

  // --- seed autoregressive x with f16 copy of generated[:, 511, :] ---
  k_yinit<<<(B_*D_+255)/256,256,0,stream>>>(outGen, yb[0]);

  // --- autoregressive loop (128 steps): cell0 -> cell1 -> projection ---
  for(int i=0; i<PRED_-S_; i++){
    const _Float16* h0p = i ? h0b[(i+1)&1] : out0 + (long)(S_-1)*H_;
    long ldh0 = i ? (long)H_ : (long)S_*H_;
    k_lstm_step<<<32,128,0,stream>>>(yb[i&1], (long)D_, D_,
        h0p, ldh0, H_, w_ih0, w_hh0, bias0, c0, h0b[i&1], (long)H_);

    const _Float16* h1p = i ? h1b[(i+1)&1] : out1 + (long)(S_-1)*H_;
    long ldh1 = i ? (long)H_ : (long)S_*H_;
    k_lstm_step<<<32,128,0,stream>>>(h0b[i&1], (long)H_, H_,
        h1p, ldh1, H_, w_ih1, w_hh1, bias1, c1, h1b[i&1], (long)H_);

    // y = h1 @ Wg^T + bg -> generated[:, 512+i, :] (f32) and yb[next] (f16)
    k_gemm<<<2,128,0,stream>>>(h1b[i&1], (long)H_, wgh, H_, bg,
        outGen, 1L, (long)PRED_*D_, 0L, (long)(S_+i)*D_, yb[(i+1)&1], B_, D_, 2);
  }
}
